// ARCUnstacked_27187142983877
// MI455X (gfx1250) — matrix operand order repaired and checked by reading, not compiled, not run on hardware
//
#include <hip/hip_runtime.h>
#include <hip/hip_bf16.h>
#include <math.h>

// ---------------- problem constants ----------------
#define BB   2
#define SEQ  4096
#define ST   128
#define DIN  1024
#define DK   64
#define DV   64
#define TT   (SEQ + 2*ST)        // 4352
#define NC   (TT/32)             // 136 key chunks of 32
#define NT16 (TT/16)             // 272 query tiles per batch
#define WAVES 4

typedef _Float16 v16h __attribute__((ext_vector_type(16)));
typedef _Float16 h8   __attribute__((ext_vector_type(8)));
typedef float    v8f  __attribute__((ext_vector_type(8)));

#define WMMA_F16(A, Bv, C) \
  __builtin_amdgcn_wmma_f32_16x16x32_f16(false, (A), false, (Bv), (short)0, (C), false, false)

// ---------------- helpers ----------------
// fast sigmoid: v_exp_f32 + v_rcp_f32 (avoid precise-div expansion)
__device__ __forceinline__ float sigmoidf_(float x) {
  return __builtin_amdgcn_rcpf(1.f + __expf(-x));
}

// DPP16 row_shr add step: v += (lane+d < 16 ? v[lane+d] : 0)
template <int CTRL>
__device__ __forceinline__ float dpp_add_step(float v) {
  int t = __builtin_amdgcn_update_dpp(0, __builtin_bit_cast(int, v),
                                      CTRL, 0xf, 0xf, true);
  return v + __builtin_bit_cast(float, t);
}
// inclusive suffix sum across 16-lane group (lane i = sum of i..15; lane0 = total)
__device__ __forceinline__ float dpp_suffix_sum16(float v) {
  v = dpp_add_step<0x111>(v);   // row_shr:1
  v = dpp_add_step<0x112>(v);   // row_shr:2
  v = dpp_add_step<0x114>(v);   // row_shr:4
  v = dpp_add_step<0x118>(v);   // row_shr:8
  return v;
}
// suffix max (lane0 = group max); invalid lanes keep own value (bound_ctrl=0)
template <int CTRL>
__device__ __forceinline__ float dpp_max_step(float v) {
  int iv = __builtin_bit_cast(int, v);
  int t = __builtin_amdgcn_update_dpp(iv, iv, CTRL, 0xf, 0xf, false);
  return fmaxf(v, __builtin_bit_cast(float, t));
}
__device__ __forceinline__ float dpp_suffix_max16(float v) {
  v = dpp_max_step<0x111>(v);
  v = dpp_max_step<0x112>(v);
  v = dpp_max_step<0x114>(v);
  v = dpp_max_step<0x118>(v);
  return v;
}
// broadcast lane0 of each 16-lane row: v_permlane16_b32 with all-zero selectors
__device__ __forceinline__ float bcast16_lane0(float v) {
  int iv = __builtin_bit_cast(int, v);
  int r = __builtin_amdgcn_permlane16(iv, iv, 0, 0, false, false);
  return __builtin_bit_cast(float, r);
}

// 16 contiguous halves -> B fragment slice (two b128 loads)
__device__ __forceinline__ v16h load_h16(const _Float16* p) {
  h8 lo = *(const h8*)p;
  h8 hi = *(const h8*)(p + 8);
  v16h r;
#pragma unroll
  for (int e = 0; e < 8; ++e) { r[e] = lo[e]; r[8 + e] = hi[e]; }
  return r;
}
// A fragment: two contiguous 16B chunks (k0..k0+7, k0+8..k0+15)
__device__ __forceinline__ v16h load_afrag(const _Float16* p, int k0) {
  h8 lo = *(const h8*)(p + k0);
  h8 hi = *(const h8*)(p + k0 + 8);
  v16h r;
#pragma unroll
  for (int e = 0; e < 8; ++e) { r[e] = lo[e]; r[8 + e] = hi[e]; }
  return r;
}

// ---------------- kernel 1: QKV projection + l2norm ----------------
__global__ __launch_bounds__(256)
void proj_kernel(const float* __restrict__ x,
                 const float* __restrict__ Wq, const float* __restrict__ Wk,
                 const float* __restrict__ Wv, const float* __restrict__ Wqs,
                 const float* __restrict__ Wks, const float* __restrict__ Wvs,
                 float* __restrict__ qf, _Float16* __restrict__ qh,
                 _Float16* __restrict__ kh, _Float16* __restrict__ vTh)
{
  __shared__ float xs[DIN];
  __shared__ float vals[192];
  __shared__ float norms[3];
  const int rt = blockIdx.x;            // 0 .. B*T-1
  const int b  = rt / TT;
  const int t  = rt % TT;
  const float* xrow = x + (size_t)rt * DIN;
  for (int i = threadIdx.x; i < DIN; i += blockDim.x) xs[i] = xrow[i];
  __syncthreads();

  const bool useS = (t < ST) || (t >= ST + SEQ);
  const int tid = threadIdx.x;
  if (tid < 192) {
    const int g = tid / 64, d = tid % 64;
    const float* W = (g == 0) ? (useS ? Wqs : Wq)
                   : (g == 1) ? (useS ? Wks : Wk)
                              : (useS ? Wvs : Wv);
    const float4* wr = (const float4*)(W + (size_t)d * DIN);
    const float4* xr = (const float4*)xs;
    float acc = 0.f;
#pragma unroll 8
    for (int i = 0; i < DIN / 4; ++i) {
      float4 a = xr[i], w4 = wr[i];
      acc += a.x * w4.x + a.y * w4.y + a.z * w4.z + a.w * w4.w;
    }
    vals[tid] = acc;
  }
  __syncthreads();
  if (tid < 3) {
    float s = 0.f;
    for (int i = 0; i < 64; ++i) { float v = vals[tid * 64 + i]; s += v * v; }
    norms[tid] = fmaxf(sqrtf(s), 1e-12f);
  }
  __syncthreads();
  if (tid < 192) {
    const int g = tid / 64, d = tid % 64;
    const float v = vals[tid] / norms[g];
    const size_t rd = (size_t)rt * 64 + d;
    if (g == 0)      { qf[rd] = v; qh[rd] = (_Float16)v; }
    else if (g == 1) { kh[rd] = (_Float16)v; }
    else             { vTh[((size_t)b * 64 + d) * TT + t] = (_Float16)v; }
  }
}

// ---------------- kernel 2: transpose cope_emb (DK,T) -> copeT (T,DK) ----------
__global__ __launch_bounds__(256)
void copet_kernel(const float* __restrict__ cope, float* __restrict__ copeT)
{
  int idx = blockIdx.x * blockDim.x + threadIdx.x;
  if (idx < TT * DK) {
    int t = idx / DK, d = idx % DK;
    copeT[idx] = cope[(size_t)d * TT + t];
  }
}

// ---------------- kernel 3: fused CoPE attention ----------------
__global__ __launch_bounds__(128)
void attn_kernel(const float* __restrict__ qf, const _Float16* __restrict__ qh,
                 const _Float16* __restrict__ kh, const _Float16* __restrict__ vTh,
                 const float* __restrict__ copeT, const float* __restrict__ scale_p,
                 float* __restrict__ out)
{
  __shared__ float    s_q[WAVES][16 * 64];      // q tile fp32 (for cope gathers)
  __shared__ _Float16 s_p[WAVES][16 * 32];      // P tile f16 (C -> A relayout)
  __shared__ float    s_carry[WAVES][NC * 16];  // [chunk][row] suffix carries

  const int w    = threadIdx.x >> 5;
  const int lane = threadIdx.x & 31;
  const int lh   = lane & 15;
  const bool hig = lane >= 16;
  const int tile = blockIdx.x * WAVES + w;
  const int b    = tile / NT16;
  const int m0   = (tile % NT16) * 16;
  const float scale = scale_p[0];

  // q fp32 tile -> LDS
  const float* qbase = qf + ((size_t)(b * TT + m0)) * 64;
  for (int i = lane; i < 16 * 64; i += 32) s_q[w][i] = qbase[i];

  // q A-fragments (K 0-31, 32-63)
  const _Float16* qhrow = qh + ((size_t)(b * TT + m0 + lh)) * 64;
  const int ak0 = hig ? 16 : 0;
  const v16h a0 = load_afrag(qhrow, ak0);
  const v16h a1 = load_afrag(qhrow + 32, ak0);

  const _Float16* kB = kh + (size_t)b * TT * 64;
  const _Float16* vB = vTh + (size_t)b * 64 * TT;
  const int koff = hig ? 16 : 0;

  // ---------- Pass A: descending sweep over ALL keys -> suffix carries ----------
  // R[r] is only meaningful in lane 0 of each 16-lane group (the carry writer).
  float R[8];
#pragma unroll
  for (int r = 0; r < 8; ++r) R[r] = 0.f;

  for (int j = NC - 1; j >= 0; --j) {
    const int s0 = j * 32;
    const _Float16* kr0 = kB + (size_t)(s0 + lh) * 64 + koff;
    const _Float16* kr1 = kB + (size_t)(s0 + 16 + lh) * 64 + koff;
    if (j > 0) __builtin_prefetch(kr0 - 32 * 64, 0, 3);   // global_prefetch_b8
    v8f c0, c1;
#pragma unroll
    for (int r = 0; r < 8; ++r) { c0[r] = 0.f; c1[r] = 0.f; }
    c0 = WMMA_F16(a0, load_h16(kr0),      c0);
    c0 = WMMA_F16(a1, load_h16(kr0 + 32), c0);
    c1 = WMMA_F16(a0, load_h16(kr1),      c1);
    c1 = WMMA_F16(a1, load_h16(kr1 + 32), c1);
    // single exec-guard: lane0 of each half writes its 8 contiguous carries,
    // then accumulates this chunk's row sums into R.
    if (lh == 0) {
      float* cw = &s_carry[w][j * 16 + (hig ? 8 : 0)];
#pragma unroll
      for (int r = 0; r < 8; ++r) cw[r] = R[r];
    }
#pragma unroll
    for (int r = 0; r < 8; ++r) {
      float g = sigmoidf_(c0[r]) + sigmoidf_(c1[r]);
      R[r] += dpp_suffix_sum16(g);                // lane0 holds chunk row sum
    }
  }

  // ---------- Pass B: causal sweep with online softmax ----------
  float m_[8], l_[8];                 // l_ valid only in lane0 of each group
  v8f o[4];
#pragma unroll
  for (int r = 0; r < 8; ++r) { m_[r] = -INFINITY; l_[r] = 0.f; }
#pragma unroll
  for (int nt = 0; nt < 4; ++nt)
#pragma unroll
    for (int r = 0; r < 8; ++r) o[nt][r] = 0.f;

  const int jmax = (m0 + 15) / 32;
  for (int j = 0; j <= jmax; ++j) {
    const int s0 = j * 32;
    const _Float16* kr0 = kB + (size_t)(s0 + lh) * 64 + koff;
    const _Float16* kr1 = kB + (size_t)(s0 + 16 + lh) * 64 + koff;
    if (j < jmax) __builtin_prefetch(kr0 + 32 * 64, 0, 3);
    v8f c0, c1;
#pragma unroll
    for (int r = 0; r < 8; ++r) { c0[r] = 0.f; c1[r] = 0.f; }
    c0 = WMMA_F16(a0, load_h16(kr0),      c0);
    c0 = WMMA_F16(a1, load_h16(kr0 + 32), c0);
    c1 = WMMA_F16(a0, load_h16(kr1),      c1);
    c1 = WMMA_F16(a1, load_h16(kr1 + 32), c1);

    // gates + in-chunk suffix scan (32 cols: tile1 then tile0+tot1)
    float suf0[8], suf1[8];
#pragma unroll
    for (int r = 0; r < 8; ++r) {
      float g1v = dpp_suffix_sum16(sigmoidf_(c1[r]));
      suf1[r] = g1v;
      float tot1 = bcast16_lane0(g1v);
      suf0[r] = dpp_suffix_sum16(sigmoidf_(c0[r])) + tot1;
    }

    // scores (logit*scale + cope_bias), masked -> -inf
    float sc0[8], sc1[8];
#pragma unroll
    for (int ct = 0; ct < 2; ++ct) {
      const int s = s0 + ct * 16 + lh;
#pragma unroll
      for (int r = 0; r < 8; ++r) {
        const int row = r + (hig ? 8 : 0);
        const int t   = m0 + row;
        const bool ok = (s <= t) && !((t >= SEQ + ST) && (s < ST));
        float sc = -INFINITY;
        if (ok) {
          float pos = (ct ? suf1[r] : suf0[r]) + s_carry[w][j * 16 + row];
          pos = fminf(pos, (float)(TT - 1));
          const float pf = floorf(pos);
          const float wv = pos - pf;
          const int ipf = (int)pf;
          const int ipc = (int)ceilf(pos);
          const float4* qr  = (const float4*)(&s_q[w][row * 64]);
          const float4* cfp = (const float4*)(copeT + (size_t)ipf * 64);
          const float4* ccp = (const float4*)(copeT + (size_t)ipc * 64);
          float af = 0.f, ac = 0.f;
#pragma unroll
          for (int i = 0; i < 16; ++i) {
            float4 qv = qr[i], xa = cfp[i], xb = ccp[i];
            af += qv.x * xa.x + qv.y * xa.y + qv.z * xa.z + qv.w * xa.w;
            ac += qv.x * xb.x + qv.y * xb.y + qv.z * xb.z + qv.w * xb.w;
          }
          const float cb = ac * wv + af * (1.f - wv);
          sc = (ct ? c1[r] : c0[r]) * scale + cb;
        }
        (ct ? sc1 : sc0)[r] = sc;
      }
    }

    // online softmax update + stage P into LDS
    float alpha[8];
#pragma unroll
    for (int r = 0; r < 8; ++r) {
      float cm = bcast16_lane0(dpp_suffix_max16(fmaxf(sc0[r], sc1[r])));
      float mn = fmaxf(m_[r], cm);
      float al = (mn == -INFINITY) ? 1.f : __expf(m_[r] - mn);
      float p0 = (sc0[r] == -INFINITY) ? 0.f : __expf(sc0[r] - mn);
      float p1 = (sc1[r] == -INFINITY) ? 0.f : __expf(sc1[r] - mn);
      l_[r] = l_[r] * al + dpp_suffix_sum16(p0 + p1);  // lane0 correct
      m_[r] = mn;
      alpha[r] = al;
      const int row = r + (hig ? 8 : 0);
      s_p[w][row * 32 + lh]      = (_Float16)p0;
      s_p[w][row * 32 + 16 + lh] = (_Float16)p1;
    }
#pragma unroll
    for (int nt = 0; nt < 4; ++nt)
#pragma unroll
      for (int r = 0; r < 8; ++r) o[nt][r] *= alpha[r];

    // P (C layout) -> A fragment via LDS; same-wave DS ops are in-order
    const v16h ap = load_afrag(&s_p[w][lh * 32], ak0);

    // o += P @ V  (vT rows give contiguous B fragments)
#pragma unroll
    for (int nt = 0; nt < 4; ++nt) {
      const _Float16* vr = vB + (size_t)(nt * 16 + lh) * TT + s0 + koff;
      o[nt] = WMMA_F16(ap, load_h16(vr), o[nt]);
    }
  }

  // epilogue: broadcast l from lane0, fast rcp, write out (B,T,64) fp32
#pragma unroll
  for (int r = 0; r < 8; ++r) {
    const int row = r + (hig ? 8 : 0);
    const float ltot = bcast16_lane0(l_[r]);
    const float inv = __builtin_amdgcn_rcpf(ltot);
    float* orow = out + ((size_t)(b * TT + m0 + row)) * 64;
#pragma unroll
    for (int nt = 0; nt < 4; ++nt) orow[nt * 16 + lh] = o[nt][r] * inv;
  }
}

// ---------------- launcher ----------------
extern "C" void kernel_launch(void* const* d_in, const int* in_sizes, int n_in,
                              void* d_out, int out_size, void* d_ws, size_t ws_size,
                              hipStream_t stream)
{
  const float* x    = (const float*)d_in[0];
  const float* Wq   = (const float*)d_in[1];
  const float* Wk   = (const float*)d_in[2];
  const float* Wv   = (const float*)d_in[3];
  const float* Wqs  = (const float*)d_in[4];
  const float* Wks  = (const float*)d_in[5];
  const float* Wvs  = (const float*)d_in[6];
  const float* cope = (const float*)d_in[7];
  const float* scl  = (const float*)d_in[8];
  float* out = (float*)d_out;

  // workspace layout (all 256B aligned)
  constexpr size_t N_ROWS = (size_t)BB * TT;          // 8704
  constexpr size_t OFF_QF   = 0;                      // fp32 q  : N_ROWS*64*4
  constexpr size_t OFF_QH   = OFF_QF + N_ROWS * 64 * 4;
  constexpr size_t OFF_KH   = OFF_QH + N_ROWS * 64 * 2;
  constexpr size_t OFF_VT   = OFF_KH + N_ROWS * 64 * 2;
  constexpr size_t OFF_CT   = OFF_VT + N_ROWS * 64 * 2;
  char* ws = (char*)d_ws;
  float*    qf   = (float*)(ws + OFF_QF);
  _Float16* qh   = (_Float16*)(ws + OFF_QH);
  _Float16* kh   = (_Float16*)(ws + OFF_KH);
  _Float16* vTh  = (_Float16*)(ws + OFF_VT);
  float*    copT = (float*)(ws + OFF_CT);

  proj_kernel<<<(int)N_ROWS, 256, 0, stream>>>(x, Wq, Wk, Wv, Wqs, Wks, Wvs,
                                               qf, qh, kh, vTh);
  copet_kernel<<<(TT * DK + 255) / 256, 256, 0, stream>>>(cope, copT);
  attn_kernel<<<(BB * NT16) / WAVES, WAVES * 32, 0, stream>>>(qf, qh, kh, vTh,
                                                              copT, scl, out);
}
